// GraphSAGEMuleDetector_764504178985
// MI455X (gfx1250) — matrix operand order repaired and checked by reading, not compiled, not run on hardware
//
#include <hip/hip_runtime.h>
#include <hip/hip_bf16.h>
#include <math.h>

// ---------------------------------------------------------------------------
// GraphSAGE 3-layer forward for MI455X (gfx1250, wave32, WMMA).
//   layer: h' = (mean_agg(h) @ Wl.T) + (h @ Wr.T) + b   [+ ReLU for L0,L1]
//   out   = sigmoid(h2 @ Wc.T + bc)
// Aggregation = fp32 atomics into an L2-resident msg buffer (102.4MB < 192MB L2).
// GEMMs      = bf16 WMMA (v_wmma_f32_16x16x32_bf16), fp32 accumulate,
//              16x64 output per wave (4 N-tiles register-blocked, 8 WMMA/k-step).
// ---------------------------------------------------------------------------

#define N_NODES 100000
#define F_DIM   256          // input / hidden width (all GEMM K = 256)

typedef __bf16 bf16_t;
typedef __attribute__((ext_vector_type(16))) __bf16 v16bf;
typedef __attribute__((ext_vector_type(8)))  float  v8f;

// ---------------- zeroing (graph-capture-safe, no hipMemset) ----------------
__global__ void zero_f4_kernel(float4* __restrict__ p, long n4) {
    long i = (long)blockIdx.x * blockDim.x + threadIdx.x;
    if (i < n4) p[i] = make_float4(0.f, 0.f, 0.f, 0.f);
}

// ---------------- degree + reciprocal degree --------------------------------
__global__ void deg_kernel(const int* __restrict__ dst, float* __restrict__ deg, int E) {
    int e = blockIdx.x * blockDim.x + threadIdx.x;
    if (e < E) atomicAdd(&deg[dst[e]], 1.0f);
}

__global__ void rdeg_kernel(const float* __restrict__ deg, float* __restrict__ rdeg, int n) {
    int i = blockIdx.x * blockDim.x + threadIdx.x;
    if (i < n) rdeg[i] = 1.0f / fmaxf(deg[i], 1.0f);
}

// ---------------- edge scatter: msg[dst] += feat[src] -----------------------
// 64 threads per edge, float4 per thread (256 feats). Atomics resolve in L2.
__global__ void scatter_sum_kernel(const float* __restrict__ feat,
                                   const int* __restrict__ src,
                                   const int* __restrict__ dst,
                                   float* __restrict__ msg, int E) {
    long tid = (long)blockIdx.x * blockDim.x + threadIdx.x;
    long total = (long)E * (F_DIM / 4);
    if (tid >= total) return;
    int e = (int)(tid >> 6);           // F_DIM/4 == 64
    int c = ((int)tid & 63) << 2;
    int s = src[e], d = dst[e];
    float4 v = *(const float4*)(feat + (size_t)s * F_DIM + c);
    float* p = msg + (size_t)d * F_DIM + c;
    atomicAdd(p + 0, v.x);
    atomicAdd(p + 1, v.y);
    atomicAdd(p + 2, v.z);
    atomicAdd(p + 3, v.w);
}

// -------- bf16 fragment pack helpers (lower to v_cvt_pk_bf16_f32) -----------
__device__ __forceinline__ void pack8(v16bf& d, int base, float4 a, float4 b) {
    d[base + 0] = (bf16_t)a.x; d[base + 1] = (bf16_t)a.y;
    d[base + 2] = (bf16_t)a.z; d[base + 3] = (bf16_t)a.w;
    d[base + 4] = (bf16_t)b.x; d[base + 5] = (bf16_t)b.y;
    d[base + 6] = (bf16_t)b.z; d[base + 7] = (bf16_t)b.w;
}
__device__ __forceinline__ void pack8s(v16bf& d, int base, float4 a, float4 b, float s) {
    d[base + 0] = (bf16_t)(a.x * s); d[base + 1] = (bf16_t)(a.y * s);
    d[base + 2] = (bf16_t)(a.z * s); d[base + 3] = (bf16_t)(a.w * s);
    d[base + 4] = (bf16_t)(b.x * s); d[base + 5] = (bf16_t)(b.y * s);
    d[base + 6] = (bf16_t)(b.z * s); d[base + 7] = (bf16_t)(b.w * s);
}

// ---------------- fused SAGE GEMM: out = (msg*rdeg)@Wl.T + root@Wr.T + b ----
// One wave per 16x64 output block (4 accumulators). K = 256 compile-time.
// Per 32-wide k-step: 2 shared A frags + 8 B frags -> 8 back-to-back WMMAs.
// A frag (lane L): row m = L%16, 16 contiguous K starting at ka=(L>=16)*16
// B frag (lane L): col n = L%16, 16 contiguous K starting at kb=(L>=16)*16
//                  == contiguous slice of W[n, k0+kb .. k0+kb+15]
// C/D    (lane L): col n = L%16, VGPR v -> row m = v + 8*(L>=16)
__global__ __launch_bounds__(256)
void sage_gemm_wmma_kernel(const float* __restrict__ msg,
                           const float* __restrict__ rdeg,
                           const float* __restrict__ root,
                           const float* __restrict__ Wl,
                           const float* __restrict__ Wr,
                           const float* __restrict__ bias,
                           float* __restrict__ out,
                           int M, int Nout, int relu) {
    const int lane    = threadIdx.x & 31;
    const int wave    = threadIdx.x >> 5;
    const int nblocks = Nout >> 6;                 // 64-wide N blocks
    const long gw     = (long)blockIdx.x * 8 + wave;
    const long wtot   = (long)(M >> 4) * nblocks;
    if (gw >= wtot) return;                        // wave-uniform: EXEC all-1s

    const int nb = (int)(gw % nblocks);
    const int mt = (int)(gw / nblocks);
    const int m  = (mt << 4) + (lane & 15);
    const int n0 = (nb << 6) + (lane & 15);        // this lane's column, j-th tile at n0+16j
    const int ka = (lane >> 4) << 4;               // 0 or 16 (A-fragment K base)
    const int kb = (lane >> 4) << 4;               // 0 or 16 (B-fragment K base)

    const float rd = rdeg[m];
    const float* aggRow  = msg  + (size_t)m * F_DIM;
    const float* rootRow = root + (size_t)m * F_DIM;
    const float* wlRow[4];
    const float* wrRow[4];
    #pragma unroll
    for (int j = 0; j < 4; ++j) {
        wlRow[j] = Wl + (size_t)(n0 + 16 * j) * F_DIM;
        wrRow[j] = Wr + (size_t)(n0 + 16 * j) * F_DIM;
    }

    v8f acc[4] = {{}, {}, {}, {}};

    #pragma unroll
    for (int k0 = 0; k0 < F_DIM; k0 += 32) {
        // prefetch next k-step of the streamed A operands (WGP-scope)
        __builtin_prefetch((const void*)(aggRow  + k0 + 32), 0, 3);
        __builtin_prefetch((const void*)(rootRow + k0 + 32), 0, 3);

        // shared A fragments (agg path scaled by 1/deg, root path raw)
        const float4* pa0 = (const float4*)(aggRow  + k0 + ka);       // k..k+7
        const float4* pa1 = (const float4*)(aggRow  + k0 + ka + 8);   // k+8..k+15
        const float4* pr0 = (const float4*)(rootRow + k0 + ka);
        const float4* pr1 = (const float4*)(rootRow + k0 + ka + 8);
        v16bf aAgg, aRoot;
        pack8s(aAgg, 0, pa0[0], pa0[1], rd);
        pack8s(aAgg, 8, pa1[0], pa1[1], rd);
        pack8 (aRoot, 0, pr0[0], pr0[1]);
        pack8 (aRoot, 8, pr1[0], pr1[1]);

        // 8 B fragments (4 Wl tiles + 4 Wr tiles), 16 contiguous K each
        v16bf bL[4], bR[4];
        #pragma unroll
        for (int j = 0; j < 4; ++j) {
            const float4* pl = (const float4*)(wlRow[j] + k0 + kb);
            const float4* pw = (const float4*)(wrRow[j] + k0 + kb);
            pack8(bL[j], 0, pl[0], pl[1]); pack8(bL[j], 8, pl[2], pl[3]);
            pack8(bR[j], 0, pw[0], pw[1]); pack8(bR[j], 8, pw[2], pw[3]);
        }

        // 8 WMMAs; same-accumulator pairs are 4 apart to avoid acc RAW stalls
        #pragma unroll
        for (int j = 0; j < 4; ++j)
            acc[j] = __builtin_amdgcn_wmma_f32_16x16x32_bf16(
                         false, aAgg, false, bL[j], (short)0, acc[j], false, false);
        #pragma unroll
        for (int j = 0; j < 4; ++j)
            acc[j] = __builtin_amdgcn_wmma_f32_16x16x32_bf16(
                         false, aRoot, false, bR[j], (short)0, acc[j], false, false);
    }

    const int mrow = (mt << 4) + ((lane >> 4) << 3);
    #pragma unroll
    for (int j = 0; j < 4; ++j) {
        const int n  = n0 + 16 * j;
        const float bv = bias[n];
        #pragma unroll
        for (int v = 0; v < 8; ++v) {
            float o = acc[j][v] + bv;
            if (relu) o = fmaxf(o, 0.0f);
            out[(size_t)(mrow + v) * Nout + n] = o;
        }
    }
}

// ---------------- classifier: sigmoid(h @ Wc.T + bc), h is [M,128] ----------
__global__ void classifier_kernel(const float* __restrict__ h,
                                  const float* __restrict__ Wc,
                                  const float* __restrict__ bc,
                                  float* __restrict__ out, int M) {
    int i = blockIdx.x * blockDim.x + threadIdx.x;
    if (i >= M) return;
    const float4* hr = (const float4*)(h + (size_t)i * 128);
    const float4* wc = (const float4*)Wc;
    float s = bc[0];
    #pragma unroll
    for (int j = 0; j < 32; ++j) {
        float4 a = hr[j], b = wc[j];
        s += a.x * b.x + a.y * b.y + a.z * b.z + a.w * b.w;
    }
    out[i] = 1.0f / (1.0f + expf(-s));
}

// ---------------------------------------------------------------------------
extern "C" void kernel_launch(void* const* d_in, const int* in_sizes, int n_in,
                              void* d_out, int out_size, void* d_ws, size_t ws_size,
                              hipStream_t stream) {
    const float* x    = (const float*)d_in[0];
    const int*   ei   = (const int*)d_in[1];
    const float* Wl0  = (const float*)d_in[2];
    const float* Wr0  = (const float*)d_in[3];
    const float* b0   = (const float*)d_in[4];
    const float* Wl1  = (const float*)d_in[5];
    const float* Wr1  = (const float*)d_in[6];
    const float* b1   = (const float*)d_in[7];
    const float* Wl2  = (const float*)d_in[8];
    const float* Wr2  = (const float*)d_in[9];
    const float* b2   = (const float*)d_in[10];
    const float* Wc   = (const float*)d_in[11];
    const float* bc   = (const float*)d_in[12];
    float*       out  = (float*)d_out;

    const int N = N_NODES;
    const int E = in_sizes[1] / 2;             // edge_index is [2, E]
    const int* src = ei;
    const int* dst = ei + E;

    // workspace carve-up (floats)
    float* ws   = (float*)d_ws;
    float* deg  = ws;                                   // N
    float* rdeg = deg  + N;                             // N
    float* msg  = rdeg + N;                             // N*256 (L2-resident)
    float* hA   = msg  + (size_t)N * F_DIM;             // N*256
    float* hB   = hA   + (size_t)N * F_DIM;             // N*256
    // layer-2 output (N*128) reuses hA (its inputs are msg + hB)

    const int T = 256;
    auto zero = [&](float* p, long nfloats) {
        long n4 = nfloats / 4;
        zero_f4_kernel<<<(int)((n4 + T - 1) / T), T, 0, stream>>>((float4*)p, n4);
    };
    auto scatter = [&](const float* feat) {
        long tot = (long)E * (F_DIM / 4);
        scatter_sum_kernel<<<(int)((tot + T - 1) / T), T, 0, stream>>>(feat, src, dst, msg, E);
    };
    auto gemm = [&](const float* root, const float* Wl, const float* Wr,
                    const float* b, float* o, int Nout, int relu) {
        long waves = (long)(N / 16) * (Nout / 64);      // one wave per 16x64 block
        sage_gemm_wmma_kernel<<<(int)((waves + 7) / 8), T, 0, stream>>>(
            msg, rdeg, root, Wl, Wr, b, o, N, Nout, relu);
    };

    // degrees (shared across layers)
    zero(deg, N);
    deg_kernel<<<(E + T - 1) / T, T, 0, stream>>>(dst, deg, E);
    rdeg_kernel<<<(N + T - 1) / T, T, 0, stream>>>(deg, rdeg, N);

    // layer 0
    zero(msg, (long)N * F_DIM);
    scatter(x);
    gemm(x, Wl0, Wr0, b0, hA, 256, 1);

    // layer 1
    zero(msg, (long)N * F_DIM);
    scatter(hA);
    gemm(hA, Wl1, Wr1, b1, hB, 256, 1);

    // layer 2 (output width 128, no relu) -> reuse hA
    zero(msg, (long)N * F_DIM);
    scatter(hB);
    gemm(hB, Wl2, Wr2, b2, hA, 128, 0);

    // classifier
    classifier_kernel<<<(N + T - 1) / T, T, 0, stream>>>(hA, Wc, bc, out, N);
}
